// multihead_attention_68822555951449
// MI455X (gfx1250) — matrix operand order repaired and checked by reading, not compiled, not run on hardware
//
#include <hip/hip_runtime.h>

#define BB 8
#define SS 1024
#define DD 1024
#define HH 16
#define EE 64
#define THRESH 0.001f

typedef __attribute__((ext_vector_type(16))) __bf16        v16bf;
typedef __attribute__((ext_vector_type(8)))  float         v8f;
typedef __attribute__((ext_vector_type(4)))  unsigned int  v4u;
typedef __attribute__((ext_vector_type(4)))  float         v4f;

union Frag16 { v16bf v; v4u q[2]; unsigned short s[16]; };
union Pack8  { v4u q; unsigned short s[8]; };

__device__ __forceinline__ unsigned short f2bf(float f) {
  unsigned int u = __float_as_uint(f);
  u += 0x7FFFu + ((u >> 16) & 1u);            // round-to-nearest-even
  return (unsigned short)(u >> 16);
}

// ---- CDNA5 async memory->LDS copy (ASYNCcnt path), per ISA §15.18.3 -------
// global_load_async_to_lds_b128: VDST = LDS byte-offset VGPR, VADDR = 64b addr.
__device__ __forceinline__ void async_copy_b128(const unsigned short* gsrc,
                                                unsigned short* ldst) {
  unsigned loff = (unsigned)(uintptr_t)ldst;                  // low 32b = LDS offset
  unsigned long long gaddr = (unsigned long long)(uintptr_t)gsrc;
  asm volatile("global_load_async_to_lds_b128 %0, %1, off"
               :: "v"(loff), "v"(gaddr) : "memory");
}
__device__ __forceinline__ void wait_async0() {
  asm volatile("s_wait_asynccnt 0x0" ::: "memory");
}

// ---------------------------------------------------------------- stage 1
__global__ void cvt_f32_bf16(const float* __restrict__ src,
                             unsigned short* __restrict__ dst, int n) {
  int i = blockIdx.x * blockDim.x + threadIdx.x;
  if (i < n) dst[i] = f2bf(src[i]);
}

// ---------------------------------------------------------------- stage 2
// One block = (h, b, 128-row tile). 8 waves, each wave computes 16 rows x 64 cols.
// A tile staged via async-to-LDS, double-buffered; W tile transposed in LDS.
// out layout: isV==0 -> [H*B][S][E] (Q,K), isV==1 -> [H*B][E][S] (V transposed).
__global__ __launch_bounds__(256)
void qkv_gemm(const unsigned short* __restrict__ xb,   // [B][S][D] bf16
              const unsigned short* __restrict__ w,    // [H][D][E] bf16
              const float* __restrict__ bias,          // [H][E]
              unsigned short* __restrict__ out,
              int isV) {
  __shared__ unsigned short lA[2][128 * 32];   // x tiles   (2 x 8 KB)
  __shared__ unsigned short lB[2][64 * 32];    // W^T tiles (2 x 4 KB)

  int blk = blockIdx.x;
  int rowblk = blk & 7, hb = blk >> 3;
  int b = hb & 7, h = hb >> 3;
  int row0 = rowblk * 128;
  int t = threadIdx.x, lane = t & 31, wave = t >> 5;
  int lm = lane & 15, lh = lane >> 4;
  const unsigned short* wh = w + (size_t)h * DD * EE;

  // per-thread staging coordinates
  int ar = t >> 1, ahalf = (t & 1) * 16;       // A: thread copies 32 B of one row
  const unsigned short* asrc = xb + ((size_t)(b * SS + row0 + ar) * DD + ahalf);
  int bk = t >> 3, be0 = (t & 7) * 8;          // B: thread scatters 8 halfs

  v8f c0 = {}, c1 = {}, c2 = {}, c3 = {};

  // prologue: stage tile 0 into buffer 0
  async_copy_b128(asrc,     &lA[0][ar * 32 + ahalf]);
  async_copy_b128(asrc + 8, &lA[0][ar * 32 + ahalf + 8]);
  {
    Pack8 p; p.q = *(const v4u*)(wh + (size_t)bk * EE + be0);
#pragma unroll
    for (int j = 0; j < 8; ++j) lB[0][(be0 + j) * 32 + bk] = p.s[j];
  }
  wait_async0();
  __syncthreads();

  for (int k0 = 0; k0 < DD; k0 += 32) {
    int buf = (k0 >> 5) & 1;
    if (k0 + 32 < DD) {                        // stage next tile while computing
      const unsigned short* srcA = asrc + (k0 + 32);
      async_copy_b128(srcA,     &lA[buf ^ 1][ar * 32 + ahalf]);
      async_copy_b128(srcA + 8, &lA[buf ^ 1][ar * 32 + ahalf + 8]);
      Pack8 p; p.q = *(const v4u*)(wh + (size_t)(k0 + 32 + bk) * EE + be0);
#pragma unroll
      for (int j = 0; j < 8; ++j) lB[buf ^ 1][(be0 + j) * 32 + bk] = p.s[j];
    }

    Frag16 a;                                  // A frag: row M, K {kb..+7, kb+16..+23}
    {
      const unsigned short* ap = &lA[buf][(wave * 16 + lm) * 32 + lh * 8];
      a.q[0] = *(const v4u*)(ap);
      a.q[1] = *(const v4u*)(ap + 16);
    }
    Frag16 bf0, bf1, bf2, bf3;                 // preload all 4 B frags -> back-to-back WMMA
    {
      const unsigned short* bp = &lB[buf][lm * 32 + lh * 8];
      bf0.q[0] = *(const v4u*)(bp);        bf0.q[1] = *(const v4u*)(bp + 16);
      bf1.q[0] = *(const v4u*)(bp + 512);  bf1.q[1] = *(const v4u*)(bp + 528);
      bf2.q[0] = *(const v4u*)(bp + 1024); bf2.q[1] = *(const v4u*)(bp + 1040);
      bf3.q[0] = *(const v4u*)(bp + 1536); bf3.q[1] = *(const v4u*)(bp + 1552);
    }
    c0 = __builtin_amdgcn_wmma_f32_16x16x32_bf16(false, a.v, false, bf0.v, (short)0, c0, false, false);
    c1 = __builtin_amdgcn_wmma_f32_16x16x32_bf16(false, a.v, false, bf1.v, (short)0, c1, false, false);
    c2 = __builtin_amdgcn_wmma_f32_16x16x32_bf16(false, a.v, false, bf2.v, (short)0, c2, false, false);
    c3 = __builtin_amdgcn_wmma_f32_16x16x32_bf16(false, a.v, false, bf3.v, (short)0, c3, false, false);

    wait_async0();                             // my staging landed in LDS
    __syncthreads();                           // everyone's staging + reads done
  }

  // epilogue: bias add, convert, store (V transposed)
#pragma unroll
  for (int nt = 0; nt < 4; ++nt) {
    v8f c = nt == 0 ? c0 : nt == 1 ? c1 : nt == 2 ? c2 : c3;
    int n = nt * 16 + lm;
    float bv = bias[h * EE + n];
#pragma unroll
    for (int i = 0; i < 8; ++i) {
      int srow = row0 + wave * 16 + i + 8 * lh;
      float val = c[i] + bv;
      size_t idx = isV ? ((size_t)(hb * EE + n) * SS + srow)
                       : ((size_t)(hb * SS + srow) * EE + n);
      out[idx] = f2bf(val);
    }
  }
}

// ---------------------------------------------------------------- stage 3
// One wave per (h,b,16-row strip). Scores strip lives in 64 KB LDS.
__global__ __launch_bounds__(32)
void attn(const unsigned short* __restrict__ qb,   // [HB][S][E] bf16
          const unsigned short* __restrict__ kmat, // [HB][S][E] bf16
          const unsigned short* __restrict__ vt,   // [HB][E][S] bf16
          const float* __restrict__ x,             // [B][S][D]
          float* __restrict__ xres,                // [B][S][D]
          float* __restrict__ att) {               // [HB][S][S]
  __shared__ float sc[16 * 1024];                  // 64 KB score strip

  int blk = blockIdx.x;
  int rt = blk & 63, hb = blk >> 6;
  int b = hb & 7, h = hb >> 3;
  int row0 = rt * 16;
  int lane = threadIdx.x;
  int lm = lane & 15, lh = lane >> 4;

  const unsigned short* qrow = qb   + (size_t)hb * SS * EE;
  const unsigned short* krow = kmat + (size_t)hb * SS * EE;
  const unsigned short* vrow = vt   + (size_t)hb * EE * SS;

  // Q fragments for both 32-wide e-steps, kept in registers all kernel
  Frag16 a0, a1;
  {
    int m = row0 + lm;
    int kb = lh * 8;
    const unsigned short* p = qrow + (size_t)m * EE;
    a0.q[0] = *(const v4u*)(p + kb);
    a0.q[1] = *(const v4u*)(p + kb + 16);
    a1.q[0] = *(const v4u*)(p + 32 + kb);
    a1.q[1] = *(const v4u*)(p + 32 + kb + 16);
  }
  const float scale = 0.03125f;  // 1/sqrt(EMBED_DIM) = 1/32

  // ---- scores = Q K^T / 32 ------------------------------------
  for (int j = 0; j < 64; ++j) {
    Frag16 b0, b1;
    int key = j * 16 + lm;
    const unsigned short* p = krow + (size_t)key * EE + lh * 8;
    if (j + 1 < 64) __builtin_prefetch(p + 16 * EE, 0, 0);   // next K tile
    b0.q[0] = *(const v4u*)(p);
    b0.q[1] = *(const v4u*)(p + 16);
    b1.q[0] = *(const v4u*)(p + 32);
    b1.q[1] = *(const v4u*)(p + 48);
    v8f c = {};
    c = __builtin_amdgcn_wmma_f32_16x16x32_bf16(false, a0.v, false, b0.v,
                                                (short)0, c, false, false);
    c = __builtin_amdgcn_wmma_f32_16x16x32_bf16(false, a1.v, false, b1.v,
                                                (short)0, c, false, false);
#pragma unroll
    for (int i = 0; i < 8; ++i)
      sc[(i + 8 * lh) * 1024 + j * 16 + lm] = c[i] * scale;
  }
  __syncthreads();

  // ---- per-row thresholded softmax -----------------------------
  for (int r = 0; r < 16; ++r) {
    float* row = sc + r * 1024;
    float m = -3.4e38f;
    for (int c0 = lane; c0 < 1024; c0 += 32) m = fmaxf(m, row[c0]);
#pragma unroll
    for (int o = 16; o > 0; o >>= 1) m = fmaxf(m, __shfl_xor(m, o, 32));
    float z = 0.f;
    for (int c0 = lane; c0 < 1024; c0 += 32) z += __expf(row[c0] - m);
#pragma unroll
    for (int o = 16; o > 0; o >>= 1) z += __shfl_xor(z, o, 32);
    // p >= thr  <=>  s >= m + log(thr * Z)
    float cut = m + __logf(THRESH * z);
    float z2 = 0.f;
    for (int c0 = lane; c0 < 1024; c0 += 32) {
      float s = row[c0];
      if (s >= cut) z2 += __expf(s - m);
    }
#pragma unroll
    for (int o = 16; o > 0; o >>= 1) z2 += __shfl_xor(z2, o, 32);
    float inv = 1.0f / (z2 + 1e-12f * z);   // = p/(sum_kept + 1e-12)
    float* arow = att + ((size_t)hb * SS + row0 + r) * SS;
    for (int c0 = lane; c0 < 1024; c0 += 32) {
      float s = row[c0];
      float p = (s >= cut) ? __expf(s - m) * inv : 0.0f;
      row[c0] = p;
      arow[c0] = p;
    }
  }
  __syncthreads();

  // ---- latent = att * V ; fuse residual add --------------------
  v8f d0 = {}, d1 = {}, d2 = {}, d3 = {};
  for (int kk = 0; kk < 1024; kk += 32) {
    Frag16 a;
    { // A fragment from f32 probability strip, converted to bf16
      const float* rp = sc + lm * 1024 + kk + lh * 8;
      v4f f0 = *(const v4f*)(rp);
      v4f f1 = *(const v4f*)(rp + 4);
      v4f f2 = *(const v4f*)(rp + 16);
      v4f f3 = *(const v4f*)(rp + 20);
#pragma unroll
      for (int i = 0; i < 4; ++i) {
        a.s[i]      = f2bf(f0[i]);
        a.s[4 + i]  = f2bf(f1[i]);
        a.s[8 + i]  = f2bf(f2[i]);
        a.s[12 + i] = f2bf(f3[i]);
      }
    }
    Frag16 bf0, bf1, bf2, bf3;               // preload all 4 V^T frags
    {
      const unsigned short* p = vrow + (size_t)lm * SS + kk + lh * 8;
      if (kk + 32 < 1024) __builtin_prefetch(p + 32, 0, 0);  // next V chunk
      bf0.q[0] = *(const v4u*)(p);             bf0.q[1] = *(const v4u*)(p + 16);
      bf1.q[0] = *(const v4u*)(p + 16 * SS);   bf1.q[1] = *(const v4u*)(p + 16 * SS + 16);
      bf2.q[0] = *(const v4u*)(p + 32 * SS);   bf2.q[1] = *(const v4u*)(p + 32 * SS + 16);
      bf3.q[0] = *(const v4u*)(p + 48 * SS);   bf3.q[1] = *(const v4u*)(p + 48 * SS + 16);
    }
    d0 = __builtin_amdgcn_wmma_f32_16x16x32_bf16(false, a.v, false, bf0.v, (short)0, d0, false, false);
    d1 = __builtin_amdgcn_wmma_f32_16x16x32_bf16(false, a.v, false, bf1.v, (short)0, d1, false, false);
    d2 = __builtin_amdgcn_wmma_f32_16x16x32_bf16(false, a.v, false, bf2.v, (short)0, d2, false, false);
    d3 = __builtin_amdgcn_wmma_f32_16x16x32_bf16(false, a.v, false, bf3.v, (short)0, d3, false, false);
  }
#pragma unroll
  for (int nt = 0; nt < 4; ++nt) {
    v8f c = nt == 0 ? d0 : nt == 1 ? d1 : nt == 2 ? d2 : d3;
    int e = nt * 16 + lm;
#pragma unroll
    for (int i = 0; i < 8; ++i) {
      int srow = row0 + i + 8 * lh;
      size_t idx = ((size_t)(b * SS + srow)) * DD + h * EE + e;
      xres[idx] = x[idx] + c[i];
    }
  }
}

// ---------------------------------------------------------------- launcher
extern "C" void kernel_launch(void* const* d_in, const int* in_sizes, int n_in,
                              void* d_out, int out_size, void* d_ws, size_t ws_size,
                              hipStream_t stream) {
  const float* x  = (const float*)d_in[0];
  const float* Wq = (const float*)d_in[1];
  const float* bq = (const float*)d_in[2];
  const float* Wk = (const float*)d_in[3];
  const float* bk = (const float*)d_in[4];
  const float* Wv = (const float*)d_in[5];
  const float* bv = (const float*)d_in[6];

  float* xres = (float*)d_out;
  float* att  = (float*)d_out + (size_t)BB * SS * DD;

  unsigned short* ws = (unsigned short*)d_ws;
  size_t o = 0;
  unsigned short* xb  = ws + o; o += (size_t)BB * SS * DD;   // x bf16
  unsigned short* wqb = ws + o; o += (size_t)HH * DD * EE;
  unsigned short* wkb = ws + o; o += (size_t)HH * DD * EE;
  unsigned short* wvb = ws + o; o += (size_t)HH * DD * EE;
  unsigned short* qb  = ws + o; o += (size_t)HH * BB * SS * EE;
  unsigned short* kb  = ws + o; o += (size_t)HH * BB * SS * EE;
  unsigned short* vt  = ws + o;                               // V transposed

  int nx = BB * SS * DD;
  int nw = HH * DD * EE;
  cvt_f32_bf16<<<dim3((nx + 255) / 256), dim3(256), 0, stream>>>(x, xb, nx);
  cvt_f32_bf16<<<dim3((nw + 255) / 256), dim3(256), 0, stream>>>(Wq, wqb, nw);
  cvt_f32_bf16<<<dim3((nw + 255) / 256), dim3(256), 0, stream>>>(Wk, wkb, nw);
  cvt_f32_bf16<<<dim3((nw + 255) / 256), dim3(256), 0, stream>>>(Wv, wvb, nw);

  dim3 gemm_grid(HH * BB * (SS / 128));   // 1024 blocks
  qkv_gemm<<<gemm_grid, dim3(256), 0, stream>>>(xb, wqb, bq, qb, 0);
  qkv_gemm<<<gemm_grid, dim3(256), 0, stream>>>(xb, wkb, bk, kb, 0);
  qkv_gemm<<<gemm_grid, dim3(256), 0, stream>>>(xb, wvb, bv, vt, 1);

  attn<<<dim3(HH * BB * (SS / 16)), dim3(32), 0, stream>>>(qb, kb, vt, x, xres, att);
}
